// LlamaQuantizedAttention_43035572306192
// MI455X (gfx1250) — 8-bit matrix operand order repaired and checked by reading, compiled, not run on hardware
//
#include <hip/hip_runtime.h>
#include <hip/hip_bf16.h>


#define B_  2
#define S_  1024
#define H_  4096
#define NH_ 32
#define NKV_ 8
#define D_  128
#define BS_ (B_ * S_)

typedef int v8i __attribute__((ext_vector_type(8)));

__device__ __forceinline__ float slot_scale(const unsigned* slot) {
    return fmaxf(__uint_as_float(*slot) / 127.0f, 1e-8f);
}

__device__ __forceinline__ void async_ld_b128(unsigned lds_addr, const void* gaddr) {
    asm volatile("global_load_async_to_lds_b128 %0, %1, off"
                 :: "v"(lds_addr), "v"((unsigned long long)(size_t)gaddr)
                 : "memory");
}
__device__ __forceinline__ void wait_async_le4() {
    asm volatile("s_wait_asynccnt 0x4" ::: "memory");
}
__device__ __forceinline__ void wait_async_0() {
    asm volatile("s_wait_asynccnt 0x0" ::: "memory");
}

__global__ void maxabs_kernel(const float* __restrict__ x, size_t n, unsigned* slot) {
    __shared__ unsigned red[256];
    unsigned m = 0u;
    for (size_t i = (size_t)blockIdx.x * blockDim.x + threadIdx.x; i < n;
         i += (size_t)gridDim.x * blockDim.x) {
        m = max(m, __float_as_uint(fabsf(x[i])));
    }
    red[threadIdx.x] = m;
    __syncthreads();
    for (int s = 128; s > 0; s >>= 1) {
        if ((int)threadIdx.x < s) red[threadIdx.x] = max(red[threadIdx.x], red[threadIdx.x + s]);
        __syncthreads();
    }
    if (threadIdx.x == 0) atomicMax(slot, red[0]);
}

__global__ void quantize_kernel(const float* __restrict__ x, signed char* __restrict__ q,
                                size_t n, const unsigned* slot) {
    float inv = 1.0f / slot_scale(slot);
    for (size_t i = (size_t)blockIdx.x * blockDim.x + threadIdx.x; i < n;
         i += (size_t)gridDim.x * blockDim.x) {
        q[i] = (signed char)(int)rintf(x[i] * inv);
    }
}

__global__ void quantize_v_kernel(const float* __restrict__ V, signed char* __restrict__ vqT,
                                  const unsigned* slot) {
    const size_t total = (size_t)BS_ * NKV_ * D_;
    size_t idx = (size_t)blockIdx.x * blockDim.x + threadIdx.x;
    if (idx >= total) return;
    float inv = 1.0f / slot_scale(slot);
    int d = idx & (D_ - 1);
    size_t r = idx >> 7;
    int h = r & (NKV_ - 1); r >>= 3;
    int s = r & (S_ - 1);
    int b = (int)(r >> 10);
    vqT[(((size_t)(b * NKV_ + h) * D_ + d) << 10) + s] = (signed char)(int)rintf(V[idx] * inv);
}

__global__ void rope_kernel(float* __restrict__ x, int nheads) {
    const size_t total = (size_t)BS_ * nheads * (D_ / 2);
    size_t idx = (size_t)blockIdx.x * blockDim.x + threadIdx.x;
    if (idx >= total) return;
    int i = idx & 63;
    size_t r = idx >> 6;
    int h = (int)(r % nheads); r /= nheads;
    int s = (int)(r & (S_ - 1));
    int b = (int)(r >> 10);
    float* p = x + (((size_t)(b * S_ + s)) * nheads + h) * D_;
    float ang = (float)s * __expf(-(float)i * (9.2103403719761840f / 64.0f));
    float sn, c;
    __sincosf(ang, &sn, &c);
    float x0 = p[i], x1 = p[i + 64];
    p[i]      = x0 * c - x1 * sn;
    p[i + 64] = x1 * c + x0 * sn;
}

__global__ __launch_bounds__(256) void gemm_i8_kernel(
    const signed char* __restrict__ A, const signed char* __restrict__ W,
    float* __restrict__ C, int M, int N, int K,
    const unsigned* sa, const unsigned* sb) {
    constexpr int LDSS = 80;
    __shared__ signed char ldsA[2][128 * LDSS];
    __shared__ signed char ldsB[2][128 * LDSS];

    const int t = threadIdx.x;
    const int lane = t & 31;
    const int wave = t >> 5;
    const int ln = lane & 15;
    const int hi = lane >> 4;
    const int wm = wave >> 1;
    const int wn = wave & 1;
    const int bm = blockIdx.y * 128;
    const int bn = blockIdx.x * 128;

    const int lrow = t >> 1;
    const int lo = (t & 1) * 32;
    const signed char* gA = A + (size_t)(bm + lrow) * K + lo;
    const signed char* gB = W + (size_t)(bn + lrow) * K + lo;
    unsigned aoff[2], boff[2];
    aoff[0] = (unsigned)(size_t)&ldsA[0][lrow * LDSS + lo];
    aoff[1] = (unsigned)(size_t)&ldsA[1][lrow * LDSS + lo];
    boff[0] = (unsigned)(size_t)&ldsB[0][lrow * LDSS + lo];
    boff[1] = (unsigned)(size_t)&ldsB[1][lrow * LDSS + lo];

    v8i acc[8];
#pragma unroll
    for (int i = 0; i < 8; i++) acc[i] = (v8i){0, 0, 0, 0, 0, 0, 0, 0};

    async_ld_b128(aoff[0], gA);
    async_ld_b128(aoff[0] + 16u, gA + 16);
    async_ld_b128(boff[0], gB);
    async_ld_b128(boff[0] + 16u, gB + 16);

    int buf = 0;
    for (int k0 = 0; k0 < K; k0 += 64, buf ^= 1) {
        if (k0 + 64 < K) {
            int nb = buf ^ 1, kn = k0 + 64;
            async_ld_b128(aoff[nb], gA + kn);
            async_ld_b128(aoff[nb] + 16u, gA + kn + 16);
            async_ld_b128(boff[nb], gB + kn);
            async_ld_b128(boff[nb] + 16u, gB + kn + 16);
            wait_async_le4();
        } else {
            wait_async_0();
        }
        __syncthreads();

        v8i afr[2], bfr[4];
#pragma unroll
        for (int mt = 0; mt < 2; mt++) {
            const signed char* ab = &ldsA[buf][(wm * 32 + mt * 16 + ln) * LDSS];
#pragma unroll
            for (int j = 0; j < 8; j++)
                afr[mt][j] = *(const int*)(ab + ((j >> 2) << 5) + (hi << 4) + ((j & 3) << 2));
        }
#pragma unroll
        for (int nt = 0; nt < 4; nt++) {
            const signed char* bb = &ldsB[buf][(wn * 64 + nt * 16 + ln) * LDSS];
#pragma unroll
            for (int j = 0; j < 8; j++)
                bfr[nt][j] = *(const int*)(bb + ((j >> 2) << 5) + (hi << 4) + ((j & 3) << 2));
        }
#pragma unroll
        for (int mt = 0; mt < 2; mt++)
#pragma unroll
            for (int nt = 0; nt < 4; nt++)
                acc[mt * 4 + nt] = __builtin_amdgcn_wmma_i32_16x16x64_iu8(
                    true, afr[mt], true, bfr[nt], acc[mt * 4 + nt], false, false);

        __syncthreads();
    }

    const float s = slot_scale(sa) * slot_scale(sb);
#pragma unroll
    for (int mt = 0; mt < 2; mt++)
#pragma unroll
        for (int nt = 0; nt < 4; nt++) {
            v8i r = acc[mt * 4 + nt];
            int col = bn + wn * 64 + nt * 16 + ln;
#pragma unroll
            for (int i = 0; i < 8; i++) {
                int row = bm + wm * 32 + mt * 16 + i + (hi << 3);
                C[(size_t)row * N + col] = (float)r[i] * s;
            }
        }
}

__global__ __launch_bounds__(256) void attn_kernel(
    const signed char* __restrict__ qq, const signed char* __restrict__ kq,
    const signed char* __restrict__ vqT, float* __restrict__ Of,
    const unsigned* __restrict__ slots) {
    const int lane = threadIdx.x & 31;
    const int wave = threadIdx.x >> 5;
    const int ln = lane & 15;
    const int hi = lane >> 4;
    const int h = blockIdx.y;
    const int b = blockIdx.z;
    const int kvh = h >> 2;
    const int qt = blockIdx.x * 128 + wave * 16;

    const float sq = slot_scale(slots + 5);
    const float sk = slot_scale(slots + 6);
    const float sv = slot_scale(slots + 7);
    const float sscale = sq * sk * 0.08838834764831845f;

    const signed char* qbase = qq + (((size_t)(b * S_ + qt + ln)) * NH_ + h) * D_;
    v8i a0, a1;
#pragma unroll
    for (int j = 0; j < 8; j++) {
        int kb = ((j >> 2) << 5) + (hi << 4) + ((j & 3) << 2);
        a0[j] = *(const int*)(qbase + kb);
        a1[j] = *(const int*)(qbase + 64 + kb);
    }

    const signed char* kbase = kq + (((size_t)(b * S_)) * NKV_ + kvh) * D_;
    const int lastcol = qt + 15;

    float rmax[8], rsum[8];
#pragma unroll
    for (int i = 0; i < 8; i++) { rmax[i] = -1e30f; rsum[i] = 0.0f; }

    for (int c = 0; c <= lastcol; c += 16) {
        const signed char* kp = kbase + (size_t)(c + ln) * (NKV_ * D_);
        v8i b0, b1;
#pragma unroll
        for (int j = 0; j < 8; j++) {
            int kb = ((j >> 2) << 5) + (hi << 4) + ((j & 3) << 2);
            b0[j] = *(const int*)(kp + kb);
            b1[j] = *(const int*)(kp + 64 + kb);
        }
        v8i acc = {0, 0, 0, 0, 0, 0, 0, 0};
        acc = __builtin_amdgcn_wmma_i32_16x16x64_iu8(true, a0, true, b0, acc, false, false);
        acc = __builtin_amdgcn_wmma_i32_16x16x64_iu8(true, a1, true, b1, acc, false, false);
#pragma unroll
        for (int i = 0; i < 8; i++) {
            int row = qt + i + (hi << 3);
            int col = c + ln;
            float f = (float)acc[i] * sscale + ((col > row) ? -1e9f : 0.0f);
            float tm = f;
#pragma unroll
            for (int m = 1; m < 16; m <<= 1) tm = fmaxf(tm, __shfl_xor(tm, m, 16));
            float p = __expf(f - tm);
#pragma unroll
            for (int m = 1; m < 16; m <<= 1) p += __shfl_xor(p, m, 16);
            float nm = fmaxf(rmax[i], tm);
            rsum[i] = rsum[i] * __expf(rmax[i] - nm) + p * __expf(tm - nm);
            rmax[i] = nm;
        }
    }

    __shared__ int plds_i[8][256];
    unsigned char* pb = (unsigned char*)plds_i[wave];

    v8i accv[8];
#pragma unroll
    for (int n = 0; n < 8; n++) accv[n] = (v8i){0, 0, 0, 0, 0, 0, 0, 0};

    const signed char* vbase = vqT + ((size_t)(b * NKV_ + kvh) * D_) * S_;

    for (int c0 = 0; c0 <= lastcol; c0 += 64) {
#pragma unroll
        for (int t = 0; t < 4; t++) {
            int c = c0 + t * 16;
            const signed char* kp = kbase + (size_t)(c + ln) * (NKV_ * D_);
            v8i b0, b1;
#pragma unroll
            for (int j = 0; j < 8; j++) {
                int kb = ((j >> 2) << 5) + (hi << 4) + ((j & 3) << 2);
                b0[j] = *(const int*)(kp + kb);
                b1[j] = *(const int*)(kp + 64 + kb);
            }
            v8i acc = {0, 0, 0, 0, 0, 0, 0, 0};
            acc = __builtin_amdgcn_wmma_i32_16x16x64_iu8(true, a0, true, b0, acc, false, false);
            acc = __builtin_amdgcn_wmma_i32_16x16x64_iu8(true, a1, true, b1, acc, false, false);
#pragma unroll
            for (int i = 0; i < 8; i++) {
                int row = qt + i + (hi << 3);
                int col = c + ln;
                float f = (float)acc[i] * sscale;
                float p = (col > row) ? 0.0f : __expf(f - rmax[i]) / rsum[i];
                pb[(i + (hi << 3)) * 64 + (c - c0) + ln] = (unsigned char)(int)rintf(p * 127.0f);
            }
        }
        asm volatile("s_wait_dscnt 0" ::: "memory");

        v8i ap;
#pragma unroll
        for (int j = 0; j < 8; j++) {
            ap[j] = *(const int*)(pb + ln * 64 + ((j >> 2) << 5) + (hi << 4) + ((j & 3) << 2));
        }
#pragma unroll
        for (int n = 0; n < 8; n++) {
            const signed char* vp = vbase + (size_t)(n * 16 + ln) * S_ + c0;
            v8i bv;
#pragma unroll
            for (int j = 0; j < 8; j++) {
                bv[j] = *(const int*)(vp + ((j >> 2) << 5) + (hi << 4) + ((j & 3) << 2));
            }
            accv[n] = __builtin_amdgcn_wmma_i32_16x16x64_iu8(true, ap, true, bv, accv[n], false, false);
        }
        asm volatile("s_wait_dscnt 0" ::: "memory");
    }

    const float oscale = sv * (1.0f / 127.0f);
#pragma unroll
    for (int n = 0; n < 8; n++) {
#pragma unroll
        for (int i = 0; i < 8; i++) {
            int row = qt + i + (hi << 3);
            int d = n * 16 + ln;
            Of[(((size_t)(b * S_ + row)) * NH_ + h) * D_ + d] = (float)accv[n][i] * oscale;
        }
    }
}

extern "C" void kernel_launch(void* const* d_in, const int* in_sizes, int n_in,
                              void* d_out, int out_size, void* d_ws, size_t ws_size,
                              hipStream_t stream) {
    const float* hidden = (const float*)d_in[0];
    const float* Wq = (const float*)d_in[1];
    const float* Wk = (const float*)d_in[2];
    const float* Wv = (const float*)d_in[3];
    const float* Wo = (const float*)d_in[4];

    char* ws = (char*)d_ws;
    unsigned* slots = (unsigned*)ws;
    size_t off = 256;
    signed char* Xq  = (signed char*)(ws + off); off += (size_t)BS_ * H_;
    signed char* Wqq = (signed char*)(ws + off); off += (size_t)H_ * H_;
    signed char* Wkq = (signed char*)(ws + off); off += (size_t)(NKV_ * D_) * H_;
    signed char* Wvq = (signed char*)(ws + off); off += (size_t)(NKV_ * D_) * H_;
    signed char* Woq = (signed char*)(ws + off); off += (size_t)H_ * H_;
    float* Qf = (float*)(ws + off); off += (size_t)BS_ * H_ * 4;
    float* Kf = (float*)(ws + off); off += (size_t)BS_ * NKV_ * D_ * 4;
    float* Vf = (float*)(ws + off); off += (size_t)BS_ * NKV_ * D_ * 4;
    signed char* qq8 = (signed char*)(ws + off); off += (size_t)BS_ * H_;
    signed char* kq8 = (signed char*)(ws + off); off += (size_t)BS_ * NKV_ * D_;
    signed char* vqT = (signed char*)(ws + off); off += (size_t)BS_ * NKV_ * D_;
    float* Of = (float*)(ws + off); off += (size_t)BS_ * H_ * 4;
    signed char* oq8 = (signed char*)(ws + off); off += (size_t)BS_ * H_;

    const size_t nX = (size_t)BS_ * H_;
    const size_t nWq = (size_t)H_ * H_;
    const size_t nWkv = (size_t)(NKV_ * D_) * H_;
    const size_t nKV = (size_t)BS_ * NKV_ * D_;

    hipMemsetAsync(slots, 0, 64, stream);

    maxabs_kernel<<<1024, 256, 0, stream>>>(hidden, nX, slots + 0);
    maxabs_kernel<<<1024, 256, 0, stream>>>(Wq, nWq, slots + 1);
    maxabs_kernel<<<1024, 256, 0, stream>>>(Wk, nWkv, slots + 2);
    maxabs_kernel<<<1024, 256, 0, stream>>>(Wv, nWkv, slots + 3);
    maxabs_kernel<<<1024, 256, 0, stream>>>(Wo, nWq, slots + 4);

    quantize_kernel<<<4096, 256, 0, stream>>>(hidden, Xq, nX, slots + 0);
    quantize_kernel<<<4096, 256, 0, stream>>>(Wq, Wqq, nWq, slots + 1);
    quantize_kernel<<<4096, 256, 0, stream>>>(Wk, Wkq, nWkv, slots + 2);
    quantize_kernel<<<4096, 256, 0, stream>>>(Wv, Wvq, nWkv, slots + 3);
    quantize_kernel<<<4096, 256, 0, stream>>>(Wo, Woq, nWq, slots + 4);

    gemm_i8_kernel<<<dim3(H_ / 128, BS_ / 128), 256, 0, stream>>>(
        Xq, Wqq, Qf, BS_, H_, H_, slots + 0, slots + 1);
    gemm_i8_kernel<<<dim3(NKV_ * D_ / 128, BS_ / 128), 256, 0, stream>>>(
        Xq, Wkq, Kf, BS_, NKV_ * D_, H_, slots + 0, slots + 2);
    gemm_i8_kernel<<<dim3(NKV_ * D_ / 128, BS_ / 128), 256, 0, stream>>>(
        Xq, Wvq, Vf, BS_, NKV_ * D_, H_, slots + 0, slots + 3);

    rope_kernel<<<(unsigned)((nX / 2 + 255) / 256), 256, 0, stream>>>(Qf, NH_);
    rope_kernel<<<(unsigned)((nKV / 2 + 255) / 256), 256, 0, stream>>>(Kf, NKV_);

    maxabs_kernel<<<1024, 256, 0, stream>>>(Qf, nX, slots + 5);
    maxabs_kernel<<<1024, 256, 0, stream>>>(Kf, nKV, slots + 6);
    maxabs_kernel<<<1024, 256, 0, stream>>>(Vf, nKV, slots + 7);
    quantize_kernel<<<4096, 256, 0, stream>>>(Qf, qq8, nX, slots + 5);
    quantize_kernel<<<4096, 256, 0, stream>>>(Kf, kq8, nKV, slots + 6);
    quantize_v_kernel<<<(unsigned)((nKV + 255) / 256), 256, 0, stream>>>(Vf, vqT, slots + 7);

    attn_kernel<<<dim3(S_ / 128, NH_, B_), 256, 0, stream>>>(qq8, kq8, vqT, Of, slots);

    maxabs_kernel<<<1024, 256, 0, stream>>>(Of, nX, slots + 8);
    quantize_kernel<<<4096, 256, 0, stream>>>(Of, oq8, nX, slots + 8);
    gemm_i8_kernel<<<dim3(H_ / 128, BS_ / 128), 256, 0, stream>>>(
        oq8, Woq, (float*)d_out, BS_, H_, H_, slots + 8, slots + 4);
}
